// RelativeAttention_48326972015122
// MI455X (gfx1250) — matrix operand order repaired and checked by reading, not compiled, not run on hardware
//
#include <hip/hip_runtime.h>
#include <math.h>

typedef __attribute__((ext_vector_type(16))) __bf16 v16bf;
typedef __attribute__((ext_vector_type(8)))  __bf16 v8bf;
typedef __attribute__((ext_vector_type(4)))  __bf16 v4bf;
typedef __attribute__((ext_vector_type(2)))  __bf16 v2bf;
typedef __attribute__((ext_vector_type(8)))  float  v8f;

#define B_   2
#define S_   1024
#define D_   1024
#define H_   16
#define DH_  64
#define NROW_ (B_ * S_)   // 2048

__device__ __forceinline__ v8f wmma_bf16(v16bf a, v16bf b, v8f c) {
  return __builtin_amdgcn_wmma_f32_16x16x32_bf16(
      false, a, false, b, (short)0, c, false, false);
}

__device__ __forceinline__ v16bf join8(v8bf lo, v8bf hi) {
  return __builtin_shufflevector(lo, hi, 0, 1, 2, 3, 4, 5, 6, 7,
                                         8, 9, 10, 11, 12, 13, 14, 15);
}
// A fragment (16-bit 16x32): per lane two contiguous 8-elem runs,
// K = {8h..8h+7} and {16+8h..16+8h+7}.  p = this lane's row base.
__device__ __forceinline__ v16bf load_a_frag(const __bf16* p, int hs) {
  return join8(*(const v8bf*)(p + 8 * hs),
               *(const v8bf*)(p + 16 + 8 * hs));
}
// B fragment (32x16): per lane two contiguous 8-elem runs,
// K = {8h..8h+7} and {16+8h..16+8h+7}.
__device__ __forceinline__ v16bf load_b_frag(const __bf16* p, int hs) {
  return join8(*(const v8bf*)(p + 8 * hs),
               *(const v8bf*)(p + 16 + 8 * hs));
}
__device__ __forceinline__ v4bf cvt4(float4 u) {
  v4bf r;
  r[0] = (__bf16)u.x; r[1] = (__bf16)u.y; r[2] = (__bf16)u.z; r[3] = (__bf16)u.w;
  return r;
}
__device__ __forceinline__ float4 up4(v4bf p) {
  float4 r;
  r.x = (float)p[0]; r.y = (float)p[1]; r.z = (float)p[2]; r.w = (float)p[3];
  return r;
}
__device__ __forceinline__ v8bf cvt8(const float* p) {
  float4 u = *(const float4*)(p);
  float4 v = *(const float4*)(p + 4);
  v8bf r;
  r[0] = (__bf16)u.x; r[1] = (__bf16)u.y; r[2] = (__bf16)u.z; r[3] = (__bf16)u.w;
  r[4] = (__bf16)v.x; r[5] = (__bf16)v.y; r[6] = (__bf16)v.z; r[7] = (__bf16)v.w;
  return r;
}
__device__ __forceinline__ float dot4(float4 a, float4 b) {
  return a.x * b.x + a.y * b.y + a.z * b.z + a.w * b.w;
}

// ---------------------------------------------------------------------------
// Kernel 1: Y = x @ W for W in {Wq,Wk,Wv}; bf16 WMMA, K-step 64.
// Q (pre-scaled by 1/sqrt(DH)) and K stored [B,H,S,DH]; V stored transposed
// [B,H,DH,S] so attention's alpha@V B-fragments are contiguous.
// ---------------------------------------------------------------------------
__global__ __launch_bounds__(256) void qkv_gemm_kernel(
    const float* __restrict__ x,
    const float* __restrict__ Wq, const float* __restrict__ Wk,
    const float* __restrict__ Wv,
    __bf16* __restrict__ Qw, __bf16* __restrict__ Kw, __bf16* __restrict__ Vt)
{
  __shared__ __bf16 xs[64][72];   // 64x64 x-tile (stride 144B, 16B-aligned)
  __shared__ __bf16 wt[64][72];   // transposed W tile: wt[n][k], 64x64

  const int tid  = threadIdx.x;
  const int wave = tid >> 5;
  const int lane = tid & 31;
  const int lm = lane & 15, hs = lane >> 4;

  const int m0 = blockIdx.x * 64;   // rows of x (B*S)
  const int n0 = blockIdx.y * 64;   // cols of Y (D)
  const int which = blockIdx.z;     // 0=Q 1=K 2=V
  const float* W = (which == 0) ? Wq : (which == 1) ? Wk : Wv;

  const int msub  = (wave & 3) * 16;
  const int nsubA = (wave >> 2) * 32;

  v8f acc0 = {}; v8f acc1 = {};

  for (int k0 = 0; k0 < D_; k0 += 64) {
    #pragma unroll
    for (int t = 0; t < 4; ++t) {
      int g = tid + t * 256;               // 1024 float4 groups
      int r = g >> 4, c4 = (g & 15) * 4;
      float4 u = *(const float4*)(x + (size_t)(m0 + r) * D_ + k0 + c4);
      *(v4bf*)&xs[r][c4] = cvt4(u);
    }
    #pragma unroll
    for (int t = 0; t < 2; ++t) {
      int g = tid + t * 256;               // 512 2(k)x4(n) blocks
      int n4 = (g & 15) * 4, kb = (g >> 4) * 2;
      float4 u0 = *(const float4*)(W + (size_t)(k0 + kb) * D_ + n0 + n4);
      float4 u1 = *(const float4*)(W + (size_t)(k0 + kb + 1) * D_ + n0 + n4);
      v4bf c0 = cvt4(u0), c1 = cvt4(u1);
      #pragma unroll
      for (int j = 0; j < 4; ++j) {
        v2bf p; p[0] = c0[j]; p[1] = c1[j];
        *(v2bf*)&wt[n4 + j][kb] = p;
      }
    }
    __syncthreads();

    // hoist all 12 ds_load_b128 ahead of the 4 WMMAs (partial waits)
    v16bf a0  = load_a_frag(&xs[msub + lm][0], hs);
    v16bf a1  = load_a_frag(&xs[msub + lm][32], hs);
    v16bf b00 = load_b_frag(&wt[nsubA + lm][0], hs);
    v16bf b10 = load_b_frag(&wt[nsubA + 16 + lm][0], hs);
    v16bf b01 = load_b_frag(&wt[nsubA + lm][32], hs);
    v16bf b11 = load_b_frag(&wt[nsubA + 16 + lm][32], hs);
    acc0 = wmma_bf16(a0, b00, acc0);
    acc1 = wmma_bf16(a0, b10, acc1);
    acc0 = wmma_bf16(a1, b01, acc0);
    acc1 = wmma_bf16(a1, b11, acc1);
    __syncthreads();
  }

  const float scale = (which == 0) ? 0.125f : 1.0f;   // 1/sqrt(64) into Q
  for (int sub = 0; sub < 2; ++sub) {
    v8f acc = sub ? acc1 : acc0;
    int n  = n0 + nsubA + sub * 16 + lm;   // column in D
    int hh = n >> 6, d = n & 63;
    int mbase = m0 + msub + 8 * hs;        // rows mbase..mbase+7 (same batch)
    int bb = mbase >> 10, ss = mbase & 1023;
    if (which == 2) {
      v8bf pk;
      #pragma unroll
      for (int r = 0; r < 8; ++r) pk[r] = (__bf16)acc[r];
      *(v8bf*)(Vt + ((size_t)(bb * H_ + hh) * DH_ + d) * S_ + ss) = pk;
    } else {
      __bf16* dst = (which == 0) ? Qw : Kw;
      #pragma unroll
      for (int r = 0; r < 8; ++r)
        dst[((size_t)(bb * H_ + hh) * S_ + (ss + r)) * DH_ + d] =
            (__bf16)(acc[r] * scale);
    }
  }
}

// ---------------------------------------------------------------------------
// Kernel 2: fused relative attention for one (b,h) and one 16-row i-tile.
// ---------------------------------------------------------------------------
__global__ __launch_bounds__(256) void rel_attn_kernel(
    const __bf16* __restrict__ Qw, const __bf16* __restrict__ Kw,
    const __bf16* __restrict__ Vt,
    const float* __restrict__ aK, const float* __restrict__ aV,
    float* __restrict__ out)
{
  extern __shared__ char smem[];
  float*  escore = (float*)smem;                       // 16x1024 scores->alpha
  __bf16* alphab = (__bf16*)(smem + 16 * 1024 * 4);    // 16x1024 alpha bf16
  float*  qf     = (float*)(smem + 16 * 1024 * 4 + 16 * 1024 * 2);  // 16x64
  float*  z1buf  = qf + 16 * 64;                       // 2 x 16x64 (K-halves)
  float*  z2buf  = z1buf + 2 * 16 * 64;                // 16x64

  const int tid  = threadIdx.x;
  const int wave = tid >> 5, lane = tid & 31;
  const int lm = lane & 15, hs = lane >> 4;

  const int bh = blockIdx.x;            // 0..31
  const int i0 = blockIdx.y * 16;
  const __bf16* Qb  = Qw + (size_t)bh * S_ * DH_;
  const __bf16* Kb  = Kw + (size_t)bh * S_ * DH_;
  const __bf16* Vtb = Vt + (size_t)bh * DH_ * S_;

  {  // stage Q tile as f32 (one v4bf load + float4 store per thread)
    int off = tid * 4;
    v4bf qv = *(const v4bf*)(Qb + (size_t)(i0 + (off >> 6)) * DH_ + (off & 63));
    *(float4*)(qf + off) = up4(qv);
  }
  __syncthreads();

  // Q fragments (two K-chunks of 32 over DH=64), built from f32 LDS
  v16bf aq0 = join8(cvt8(qf + lm * 64 + 8 * hs),
                    cvt8(qf + lm * 64 + 16 + 8 * hs));
  v16bf aq1 = join8(cvt8(qf + lm * 64 + 32 + 8 * hs),
                    cvt8(qf + lm * 64 + 48 + 8 * hs));

  // ---- phase A1: raw scores e1 = q.k via WMMA ----
  for (int t = 0; t < 8; ++t) {
    int j0 = (wave * 8 + t) * 16;
    const __bf16* kp = Kb + (size_t)(j0 + lm) * DH_;   // B[k][n]=K[j0+n][k]
    v16bf b0 = load_b_frag(kp, hs);
    v16bf b1 = load_b_frag(kp + 32, hs);
    v8f acc = {};
    acc = wmma_bf16(aq0, b0, acc);
    acc = wmma_bf16(aq1, b1, acc);
    #pragma unroll
    for (int r = 0; r < 8; ++r)
      escore[(r + 8 * hs) * 1024 + j0 + lm] = acc[r];
  }
  __syncthreads();

  // ---- phase A2: e += q.aK; one thread per (row, 64-col group).
  // q row register-resident; 4 parallel contiguous 256B aK row streams.
  {
    int m  = tid >> 4;
    int c0 = (tid & 15) * 64;
    float4 qreg[16];
    #pragma unroll
    for (int d4 = 0; d4 < 16; ++d4)
      qreg[d4] = *(const float4*)(qf + m * 64 + d4 * 4);
    const float* akrow = aK + ((size_t)(i0 + m) * S_ + c0) * DH_;
    float* erow = escore + m * 1024 + c0;
    for (int jb = 0; jb < 64; jb += 4) {
      const float* p0 = akrow + (size_t)(jb + 0) * DH_;
      const float* p1 = akrow + (size_t)(jb + 1) * DH_;
      const float* p2 = akrow + (size_t)(jb + 2) * DH_;
      const float* p3 = akrow + (size_t)(jb + 3) * DH_;
      float s0 = 0.f, s1 = 0.f, s2 = 0.f, s3 = 0.f;
      #pragma unroll 4
      for (int d4 = 0; d4 < 16; ++d4) {
        float4 q = qreg[d4];
        s0 += dot4(q, *(const float4*)(p0 + d4 * 4));
        s1 += dot4(q, *(const float4*)(p1 + d4 * 4));
        s2 += dot4(q, *(const float4*)(p2 + d4 * 4));
        s3 += dot4(q, *(const float4*)(p3 + d4 * 4));
      }
      float4 e = *(const float4*)(erow + jb);
      e.x += s0; e.y += s1; e.z += s2; e.w += s3;
      *(float4*)(erow + jb) = e;
    }
  }
  __syncthreads();

  // ---- phase B: softmax over 1024-wide rows; one wave per 2 rows ----
  for (int rr = 0; rr < 2; ++rr) {
    int m = wave * 2 + rr;
    float* row = escore + m * 1024;
    float mx = -3.0e38f;
    for (int j = lane * 4; j < 1024; j += 128) {
      float4 v = *(const float4*)(row + j);
      mx = fmaxf(mx, fmaxf(fmaxf(v.x, v.y), fmaxf(v.z, v.w)));
    }
    for (int o = 16; o >= 1; o >>= 1) mx = fmaxf(mx, __shfl_xor(mx, o, 32));
    float sum = 0.f;
    for (int j = lane * 4; j < 1024; j += 128) {
      float4 v = *(const float4*)(row + j);
      v.x = __expf(v.x - mx); v.y = __expf(v.y - mx);
      v.z = __expf(v.z - mx); v.w = __expf(v.w - mx);
      *(float4*)(row + j) = v;
      sum += v.x + v.y + v.z + v.w;
    }
    for (int o = 16; o >= 1; o >>= 1) sum += __shfl_xor(sum, o, 32);
    float inv = 1.f / sum;
    for (int j = lane * 4; j < 1024; j += 128) {
      float4 v = *(const float4*)(row + j);
      v.x *= inv; v.y *= inv; v.z *= inv; v.w *= inv;
      *(float4*)(row + j) = v;                    // f32 alpha for z2
      *(v4bf*)(alphab + m * 1024 + j) = cvt4(v);  // bf16 alpha for WMMA
    }
  }
  __syncthreads();

  // ---- phase C1: z1 = alpha @ V via WMMA, ALL 8 waves.
  // wave = (n-subtile, K-half): 16 WMMAs each; partials into z1buf halves.
  {
    int n0   = (wave & 3) * 16;       // d-subtile
    int half = wave >> 2;             // K range half*512 .. +511
    v8f acc = {};
    for (int kc = half * 16; kc < half * 16 + 16; ++kc) {
      int j0 = kc * 32;
      v16bf a = load_a_frag(alphab + lm * 1024 + j0, hs);
      // B[k][n] = V[j0+k][n0+n] = Vt[n0+n][j0+k]: contiguous along k
      v16bf b = load_b_frag(Vtb + (size_t)(n0 + lm) * S_ + j0, hs);
      acc = wmma_bf16(a, b, acc);
    }
    #pragma unroll
    for (int r = 0; r < 8; ++r)
      z1buf[half * 1024 + (r + 8 * hs) * 64 + n0 + lm] = acc[r];
  }

  // ---- phase C2: z2 = alpha.aV, ALL 8 waves (2 rows each) — the aV stream
  {
    int d0 = lane * 2;          // each lane owns 2 of 64 d's (coalesced)
    for (int rr = 0; rr < 2; ++rr) {
      int m = wave * 2 + rr;
      const float* avp  = aV + ((size_t)(i0 + m) * S_) * DH_ + d0;
      const float* arow = escore + m * 1024;
      float ax = 0.f, ay = 0.f;
      int j = 0;
      for (; j < 1024 - 32; j += 4) {
        float4 al = *(const float4*)(arow + j);
        __builtin_prefetch(avp + (size_t)(j + 32) * DH_, 0, 0);
        float2 v0 = *(const float2*)(avp + (size_t)(j + 0) * DH_);
        float2 v1 = *(const float2*)(avp + (size_t)(j + 1) * DH_);
        float2 v2 = *(const float2*)(avp + (size_t)(j + 2) * DH_);
        float2 v3 = *(const float2*)(avp + (size_t)(j + 3) * DH_);
        ax += al.x * v0.x + al.y * v1.x + al.z * v2.x + al.w * v3.x;
        ay += al.x * v0.y + al.y * v1.y + al.z * v2.y + al.w * v3.y;
      }
      for (; j < 1024; j += 4) {
        float4 al = *(const float4*)(arow + j);
        float2 v0 = *(const float2*)(avp + (size_t)(j + 0) * DH_);
        float2 v1 = *(const float2*)(avp + (size_t)(j + 1) * DH_);
        float2 v2 = *(const float2*)(avp + (size_t)(j + 2) * DH_);
        float2 v3 = *(const float2*)(avp + (size_t)(j + 3) * DH_);
        ax += al.x * v0.x + al.y * v1.x + al.z * v2.x + al.w * v3.x;
        ay += al.x * v0.y + al.y * v1.y + al.z * v2.y + al.w * v3.y;
      }
      z2buf[m * 64 + d0]     = ax;
      z2buf[m * 64 + d0 + 1] = ay;
    }
  }
  __syncthreads();

  {  // combine z1 halves + z2, one float4 per thread
    int off = tid * 4;
    int m = off >> 6, d = off & 63;
    float4 a = *(const float4*)(z1buf + off);
    float4 b2 = *(const float4*)(z1buf + 1024 + off);
    float4 c = *(const float4*)(z2buf + off);
    float4 s;
    s.x = a.x + b2.x + c.x; s.y = a.y + b2.y + c.y;
    s.z = a.z + b2.z + c.z; s.w = a.w + b2.w + c.w;
    const int b = bh >> 4, h = bh & 15;
    *(float4*)(out + ((size_t)b * S_ + (i0 + m)) * D_ + h * DH_ + d) = s;
  }
}

// ---------------------------------------------------------------------------
extern "C" void kernel_launch(void* const* d_in, const int* in_sizes, int n_in,
                              void* d_out, int out_size, void* d_ws, size_t ws_size,
                              hipStream_t stream) {
  (void)in_sizes; (void)n_in; (void)out_size; (void)ws_size;
  const float* x  = (const float*)d_in[0];
  const float* Wq = (const float*)d_in[1];
  const float* Wk = (const float*)d_in[2];
  const float* Wv = (const float*)d_in[3];
  const float* aK = (const float*)d_in[4];
  const float* aV = (const float*)d_in[5];
  float* out = (float*)d_out;

  // Workspace: Q,K [B,H,S,DH] bf16; Vt [B,H,DH,S] bf16 — 4 MB each.
  __bf16* Qw = (__bf16*)d_ws;
  __bf16* Kw = Qw + (size_t)NROW_ * D_;
  __bf16* Vt = Kw + (size_t)NROW_ * D_;

  dim3 g1(NROW_ / 64, D_ / 64, 3);
  qkv_gemm_kernel<<<g1, 256, 0, stream>>>(x, Wq, Wk, Wv, Qw, Kw, Vt);

  size_t smem = (size_t)16 * 1024 * 4     // escore
              + (size_t)16 * 1024 * 2     // alphab
              + (size_t)16 * 64 * 4       // qf
              + (size_t)2 * 16 * 64 * 4   // z1 (two K-halves)
              + (size_t)16 * 64 * 4;      // z2   => 114,688 B
  dim3 g2(B_ * H_, S_ / 16);  // x fastest: blocks sharing an i-tile adjacent
  rel_attn_kernel<<<g2, 256, smem, stream>>>(Qw, Kw, Vt, aK, aV, out);
}
